// Quantizer_81355270521166
// MI455X (gfx1250) — matrix operand order repaired and checked by reading, not compiled, not run on hardware
//
#include <hip/hip_runtime.h>
#include <hip/hip_bf16.h>

typedef __attribute__((ext_vector_type(16))) _Float16 v16h;
typedef __attribute__((ext_vector_type(8)))  float    v8f;
typedef __attribute__((ext_vector_type(4)))  int      v4i;

#define DD 256
#define KK 1024
#define TT 4096
#define BB 16
#define NN (BB * TT)          // 65536 rows
#define ND (NN * DD)          // 16777216 elements of quant

#define AS1 __attribute__((address_space(1)))
#define AS3 __attribute__((address_space(3)))

#define USE_ASYNC_LDS __has_builtin(__builtin_amdgcn_global_load_async_to_lds_b128)

// ---------------------------------------------------------------------------
// Kernel 1: zero accumulators (ws is poisoned by harness each run)
// ---------------------------------------------------------------------------
__global__ void vq_init(float* sse, float* counts) {
    int id = blockIdx.x * blockDim.x + threadIdx.x;
    if (id == 0) *sse = 0.0f;
    if (id < KK) counts[id] = 0.0f;
}

// ---------------------------------------------------------------------------
// Kernel 2: pack emb[D,K] f32 -> f16 B-fragments (32x16 WMMA B layout) and
// compute cnorm[k] = ||c_k||^2.
// Fragment layout (dense 16-bit B, 32x16): lane L holds column n=L%16;
// lanes 0-15 hold K=0..15, lanes 16-31 hold K=16..31; element j: k=16*(L/16)+j.
// Packed addressing: half index = ((ct*8+dc)*32 + lane)*16 + j
// ---------------------------------------------------------------------------
__global__ void vq_pack(const float* __restrict__ emb,
                        _Float16* __restrict__ embP,
                        float* __restrict__ cnorm) {
    int id = blockIdx.x * blockDim.x + threadIdx.x;   // 0 .. 262143
    int j    = id & 15;
    int lane = (id >> 4) & 31;
    int frag = id >> 9;            // ct*8 + dc, 0..511
    int dc   = frag & 7;
    int ct   = frag >> 3;
    int group = lane >> 4;
    int n     = lane & 15;
    int kd  = 16 * group + j;      // K-dim offset within 32-chunk
    int d   = dc * 32 + kd;
    int col = ct * 16 + n;
    embP[id] = (_Float16)emb[(size_t)d * KK + col];

    if (id < KK) {
        float s = 0.0f;
        for (int dd2 = 0; dd2 < DD; ++dd2) {
            float c = emb[(size_t)dd2 * KK + id];
            s += c * c;
        }
        cnorm[id] = s;
    }
}

// ---------------------------------------------------------------------------
// Kernel 3: main VQ.
// Block = 256 threads = 8 waves; each wave owns 32 rows (two 16-row M-tiles),
// block owns 256 consecutive t within one b. The packed B codebook is staged
// chunk-by-chunk (4 col-tiles = 32 KB) into double-buffered LDS with async
// copies, so L2 traffic is paid once per 256 rows and each B fragment read
// from LDS feeds two WMMAs.
// ---------------------------------------------------------------------------
__global__ __launch_bounds__(256) void vq_main(
    const float*    __restrict__ x,
    const float*    __restrict__ emb,
    const _Float16* __restrict__ embP,
    const float*    __restrict__ cnorm,
    float*          __restrict__ counts,
    float*          __restrict__ sseAcc,
    float*          __restrict__ outQuant,
    float*          __restrict__ outInd) {

    __shared__ uint4 lbufU[2][2048];   // 2 x 32KB B chunks
    __shared__ int   sIdx[8][32];

    const int tid   = threadIdx.x;
    const int lane  = tid & 31;
    const int wave  = tid >> 5;
    const int m     = lane & 15;
    const int group = lane >> 4;

    const int b     = blockIdx.x >> 4;          // 16 blocks per b (T/256)
    const int tbase = (blockIdx.x & 15) << 8;   // 256 rows per block
    const int t0w   = tbase + wave * 32;        // this wave's 32 rows
    const int t_0   = t0w + m;                  // M-tile 0 row for this lane
    const int t_1   = t0w + 16 + m;             // M-tile 1 row

    const float* xb = x + (size_t)b * DD * TT;

    // --- load A fragments (16x32 f16, contiguous k=16*group+j, same as packed B), two M-tiles ------------
    v16h a0[8], a1[8];
#pragma unroll
    for (int dc = 0; dc < 8; ++dc) {
#pragma unroll
        for (int j = 0; j < 16; ++j) {
            int v = j >> 1, p = j & 1;
            int kA = 16 * group + j + 0 * (v + p);
            int d = dc * 32 + kA;
            a0[dc][j] = (_Float16)xb[(size_t)d * TT + t_0];
            a1[dc][j] = (_Float16)xb[(size_t)d * TT + t_1];
        }
    }

    float minv0[8], minv1[8];
    int   mini0[8], mini1[8];
#pragma unroll
    for (int r = 0; r < 8; ++r) {
        minv0[r] = 3.4e38f; mini0[r] = 0;
        minv1[r] = 3.4e38f; mini1[r] = 0;
    }

    const uint4* embPU = (const uint4*)embP;    // 32768 x 16B total

    // --- stage chunk 0 ----------------------------------------------------
#if USE_ASYNC_LDS
#pragma unroll
    for (int j = 0; j < 8; ++j) {
        __builtin_amdgcn_global_load_async_to_lds_b128(
            (v4i*)(embPU + (size_t)tid * 8 + j),
            (AS3 v4i*)(&lbufU[0][tid * 8 + j]), 0, 0);
    }
    asm volatile("s_wait_asynccnt 0" ::: "memory");
#else
    {
        uint4 tmp[8];
#pragma unroll
        for (int j = 0; j < 8; ++j) tmp[j] = embPU[(size_t)tid * 8 + j];
#pragma unroll
        for (int j = 0; j < 8; ++j) lbufU[0][tid * 8 + j] = tmp[j];
    }
#endif
    __syncthreads();

    // --- sweep 16 chunks x 4 col-tiles ------------------------------------
    for (int c = 0; c < 16; ++c) {
        const int cb = c & 1;
#if USE_ASYNC_LDS
        if (c + 1 < 16) {
#pragma unroll
            for (int j = 0; j < 8; ++j) {
                __builtin_amdgcn_global_load_async_to_lds_b128(
                    (v4i*)(embPU + (size_t)(c + 1) * 2048 + tid * 8 + j),
                    (AS3 v4i*)(&lbufU[cb ^ 1][tid * 8 + j]), 0, 0);
            }
        }
#else
        uint4 tmp[8];
        if (c + 1 < 16) {
#pragma unroll
            for (int j = 0; j < 8; ++j)
                tmp[j] = embPU[(size_t)(c + 1) * 2048 + tid * 8 + j];
        }
#endif
        const v16h* bfp = (const v16h*)lbufU[cb];
#pragma unroll
        for (int tt = 0; tt < 4; ++tt) {
            v8f acc0 = {}, acc1 = {};
#pragma unroll
            for (int dc = 0; dc < 8; ++dc) {
                v16h bf = bfp[(tt * 8 + dc) * 32 + lane];
                acc0 = __builtin_amdgcn_wmma_f32_16x16x32_f16(
                    false, a0[dc], false, bf, (short)0, acc0, false, false);
                acc1 = __builtin_amdgcn_wmma_f32_16x16x32_f16(
                    false, a1[dc], false, bf, (short)0, acc1, false, false);
            }
            const int col = (c * 4 + tt) * 16 + m;
            const float cn = cnorm[col];
#pragma unroll
            for (int r = 0; r < 8; ++r) {
                float s0 = cn - 2.0f * acc0[r];   // ||x||^2 omitted (row-const)
                if (s0 < minv0[r]) { minv0[r] = s0; mini0[r] = col; }
                float s1 = cn - 2.0f * acc1[r];
                if (s1 < minv1[r]) { minv1[r] = s1; mini1[r] = col; }
            }
        }
#if USE_ASYNC_LDS
        if (c + 1 < 16) asm volatile("s_wait_asynccnt 0" ::: "memory");
#else
        if (c + 1 < 16) {
#pragma unroll
            for (int j = 0; j < 8; ++j) lbufU[cb ^ 1][tid * 8 + j] = tmp[j];
        }
#endif
        __syncthreads();
    }

    // --- argmin reduce across the 16 lanes of each half-wave --------------
#pragma unroll
    for (int r = 0; r < 8; ++r) {
#pragma unroll
        for (int off = 8; off > 0; off >>= 1) {
            float ov = __shfl_xor(minv0[r], off, 32);
            int   oi = __shfl_xor(mini0[r], off, 32);
            if (ov < minv0[r] || (ov == minv0[r] && oi < mini0[r])) {
                minv0[r] = ov; mini0[r] = oi;
            }
            float ov1 = __shfl_xor(minv1[r], off, 32);
            int   oi1 = __shfl_xor(mini1[r], off, 32);
            if (ov1 < minv1[r] || (ov1 == minv1[r] && oi1 < mini1[r])) {
                minv1[r] = ov1; mini1[r] = oi1;
            }
        }
    }

    // lane 0 of each 16-lane group holds rows r + 8*group of its M-tile
    if (m == 0) {
#pragma unroll
        for (int r = 0; r < 8; ++r) {
            int row0 = r + 8 * group;
            sIdx[wave][row0] = mini0[r];
            atomicAdd(&counts[mini0[r]], 1.0f);
            outInd[(size_t)b * TT + t0w + row0] = (float)mini0[r];

            int row1 = 16 + r + 8 * group;
            sIdx[wave][row1] = mini1[r];
            atomicAdd(&counts[mini1[r]], 1.0f);
            outInd[(size_t)b * TT + t0w + row1] = (float)mini1[r];
        }
    }
    __syncthreads();

    // --- gather quant, write [B,D,T], accumulate SSE ----------------------
    float sse = 0.0f;
    float* ob = outQuant + (size_t)b * DD * TT;
#pragma unroll
    for (int mt = 0; mt < 2; ++mt) {
        const int t   = t0w + mt * 16 + m;
        const int idx = sIdx[wave][mt * 16 + m];
        for (int dd2 = 0; dd2 < 128; ++dd2) {
            int d = dd2 * 2 + group;
            float q  = emb[(size_t)d * KK + idx];
            float xv = xb[(size_t)d * TT + t];
            ob[(size_t)d * TT + t] = q;
            float e = q - xv;
            sse += e * e;
        }
    }
#pragma unroll
    for (int off = 16; off > 0; off >>= 1) sse += __shfl_xor(sse, off, 32);
    if (lane == 0) atomicAdd(sseAcc, sse);
}

// ---------------------------------------------------------------------------
// Kernel 4: diff + perplexity
// ---------------------------------------------------------------------------
__global__ __launch_bounds__(1024) void vq_fin(const float* __restrict__ counts,
                                               const float* __restrict__ sse,
                                               float* diffOut, float* pplOut) {
    __shared__ float red[32];
    int tid = threadIdx.x;            // 1024 threads == K
    float p = counts[tid] / (float)NN;
    float e = p * logf(p + 1e-10f);
#pragma unroll
    for (int off = 16; off > 0; off >>= 1) e += __shfl_xor(e, off, 32);
    if ((tid & 31) == 0) red[tid >> 5] = e;
    __syncthreads();
    if (tid < 32) {
        float v = red[tid];
#pragma unroll
        for (int off = 16; off > 0; off >>= 1) v += __shfl_xor(v, off, 32);
        if (tid == 0) {
            *pplOut  = expf(-v);
            *diffOut = 1.25f * (*sse) / (float)ND;
        }
    }
}

// ---------------------------------------------------------------------------
extern "C" void kernel_launch(void* const* d_in, const int* in_sizes, int n_in,
                              void* d_out, int out_size, void* d_ws, size_t ws_size,
                              hipStream_t stream) {
    const float* x   = (const float*)d_in[0];   // [16,256,4096]
    const float* emb = (const float*)d_in[1];   // [256,1024]

    float* out   = (float*)d_out;
    float* quant = out;                         // [16,256,4096]
    float* diff  = out + ND;                    // scalar
    float* ind   = out + ND + 1;                // [16,4096] as float
    float* ppl   = out + ND + 1 + NN;           // scalar

    float*    wsf    = (float*)d_ws;
    float*    sse    = wsf;                     // 1 float (slot 0 of 16)
    float*    counts = wsf + 16;                // K floats
    float*    cnorm  = wsf + 16 + KK;           // K floats
    _Float16* embP   = (_Float16*)(wsf + 16 + 2 * KK);  // D*K halfs (512 KB)

    vq_init<<<4, 256, 0, stream>>>(sse, counts);
    vq_pack<<<(DD * KK) / 256, 256, 0, stream>>>(emb, embP, cnorm);
    vq_main<<<NN / 256, 256, 0, stream>>>(x, emb, embP, cnorm,
                                          counts, sse, quant, ind);
    vq_fin<<<1, 1024, 0, stream>>>(counts, sse, diff, ppl);
}
